// ContextVAE_17162689315186
// MI455X (gfx1250) — matrix operand order repaired and checked by reading, not compiled, not run on hardware
//
#include <hip/hip_runtime.h>
#include <hip/hip_bf16.h>
#include <stdint.h>

// ---------------- types for WMMA ----------------
typedef __attribute__((ext_vector_type(16))) _Float16 v16h;
typedef __attribute__((ext_vector_type(8)))  float    v8f;

#define D_MODEL 512
#define D_MOTION 256
#define D_FF 2048
#define NHEAD 8
#define DHEAD 64
#define BB 32
#define TT 256
#define SENC 258
#define WIN 15

enum { ACT_NONE = 0, ACT_PRELU = 1, ACT_RELU = 2 };

// ================= GEMM: C = act(A[MxK] @ W[KxN] + bias) (+resid) ==============
// f32 global -> f16 LDS tiles -> v_wmma_f32_16x16x32_f16 -> f32 out.
// Block tile 64x128, 8 waves (2 along M x 4 along N), each wave 32x32 via 2x2 WMMA.
// LDS: A tile [row][k], W tile TRANSPOSED [col][k]; row stride 40 halves (80B = 20
// dwords) -> every fragment is a 16B-aligned contiguous run => ds_load_b128,
// conflict-free (20*i mod 64 distinct for 16 lanes).
// FULL: all tile bounds exact -> no guards. CN/CK: compile-time N/K (0 = runtime)
// so W-tile loads get immediate offsets (s_clause'd) and strides become shifts.
#define BM 64
#define BN 128
#define BK 32
#define KPAD 8

template <int ACT, bool FULL, int CN, int CK>
__launch_bounds__(256)
__global__ void gemm_wmma_k(const float* __restrict__ A, const float* __restrict__ W,
                            const float* __restrict__ bias, const float* __restrict__ resid,
                            const float* __restrict__ alpha_ptr, float* __restrict__ C,
                            int M, int Nr, int Kr) {
  const int N = CN ? CN : Nr;
  const int K = CK ? CK : Kr;
  __shared__ _Float16 As[BM][BK + KPAD];
  __shared__ _Float16 Bs[BN][BK + KPAD];   // transposed: [col][k]

  const int tid  = threadIdx.x;
  const int wave = tid >> 5;
  const int lane = tid & 31;
  const int wm = wave & 1;        // 0..1  (32 rows each)
  const int wn = wave >> 1;       // 0..3  (32 cols each)
  const int bm = blockIdx.x * BM;
  const int bn = blockIdx.y * BN;
  const int lrow = lane & 15;
  const int lhi  = lane >> 4;

  // A-load slot: this thread owns row ar, 8 consecutive k at ac8
  const int ar  = tid >> 2;
  const int ac8 = (tid & 3) * 8;
  v8f acc[2][2] = {};

  for (int k0 = 0; k0 < K; k0 += BK) {
    // ---- A tile: 8 contiguous floats -> cvt -> one 16B LDS chunk ----
    {
      const int gr = bm + ar;
      const float* src = A + (size_t)gr * K + (k0 + ac8);
      _Float16 h[8];
      if (FULL || (gr < M && (k0 + ac8 + 8) <= K)) {
#pragma unroll
        for (int j = 0; j < 8; ++j) h[j] = (_Float16)src[j];
      } else {
#pragma unroll
        for (int j = 0; j < 8; ++j) {
          const int gc = k0 + ac8 + j;
          h[j] = (_Float16)((gr < M && gc < K) ? src[j] : 0.0f);
        }
      }
#pragma unroll
      for (int j = 0; j < 8; ++j) As[ar][ac8 + j] = h[j];
    }
    // ---- W tile, transposed into LDS: thread reads 8 k's of one column ----
#pragma unroll
    for (int s = 0; s < 2; ++s) {
      const int slot = tid + s * 256;
      const int col  = slot & 127;
      const int kg   = slot >> 7;          // 0..3
      const int gc   = bn + col;
      const int kb   = k0 + kg * 8;
      const float* wsrc = W + (size_t)kb * N + gc;   // + j*N per element (const if CN)
      _Float16 h[8];
      if (FULL || (gc < N && (kb + 8) <= K)) {
#pragma unroll
        for (int j = 0; j < 8; ++j) h[j] = (_Float16)wsrc[(size_t)j * N];
      } else {
#pragma unroll
        for (int j = 0; j < 8; ++j)
          h[j] = (_Float16)((gc < N && (kb + j) < K) ? wsrc[(size_t)j * N] : 0.0f);
      }
#pragma unroll
      for (int j = 0; j < 8; ++j) Bs[col][kg * 8 + j] = h[j];
    }
    // prefetch next K-chunk of A (global_prefetch_b8)
    if (k0 + BK < K) {
      const int gr = bm + ar;
      if (FULL || (gr < M && (k0 + BK + ac8) < K))
        __builtin_prefetch(&A[(size_t)gr * K + (k0 + BK + ac8)], 0, 3);
    }
    __syncthreads();

    // ---- fragments (ISA 16-bit wave32 layouts), all contiguous b128 runs ----
    v16h af[2], bf[2];
#pragma unroll
    for (int t = 0; t < 2; ++t) {
      const int arow = wm * 32 + t * 16 + lrow;
      const int kh = lhi * 8;
#pragma unroll
      for (int g = 0; g < 2; ++g)
#pragma unroll
        for (int j = 0; j < 8; ++j)
          af[t][g * 8 + j] = As[arow][kh + g * 16 + j];
      const int bcol = wn * 32 + t * 16 + lrow;
      const int kb2 = lhi * 8;
#pragma unroll
      for (int g = 0; g < 2; ++g)
#pragma unroll
        for (int j = 0; j < 8; ++j)
          bf[t][g * 8 + j] = Bs[bcol][kb2 + g * 16 + j];
    }
#pragma unroll
    for (int t = 0; t < 2; ++t)
#pragma unroll
      for (int u = 0; u < 2; ++u)
        acc[t][u] = __builtin_amdgcn_wmma_f32_16x16x32_f16(
            false, af[t], false, bf[u], (short)0, acc[t][u], false, false);
    __syncthreads();
  }

  const float alpha = alpha_ptr ? alpha_ptr[0] : 0.0f;
#pragma unroll
  for (int t = 0; t < 2; ++t) {
    const int rowb = bm + wm * 32 + t * 16 + lhi * 8;  // C layout: VGPR r -> M=r / r+8
#pragma unroll
    for (int u = 0; u < 2; ++u) {
      const int col = bn + wn * 32 + u * 16 + lrow;
      if (FULL || col < N) {
        const float bv = bias ? bias[col] : 0.0f;
#pragma unroll
        for (int r = 0; r < 8; ++r) {
          const int row = rowb + r;
          if (FULL || row < M) {
            float v = acc[t][u][r] + bv;
            if (ACT == ACT_PRELU) v = (v >= 0.0f) ? v : alpha * v;
            else if (ACT == ACT_RELU) v = fmaxf(v, 0.0f);
            if (resid) v += resid[(size_t)row * N + col];
            C[(size_t)row * N + col] = v;
          }
        }
      }
    }
  }
}

// ================= LayerNorm: one wave per 512-wide row =================
__launch_bounds__(256)
__global__ void layernorm_k(const float* __restrict__ x, const float* __restrict__ g,
                            const float* __restrict__ b, float* __restrict__ y, int rows) {
  const int wave = (blockIdx.x * blockDim.x + threadIdx.x) >> 5;
  const int lane = threadIdx.x & 31;
  if (wave >= rows) return;
  const float* xr = x + (size_t)wave * D_MODEL;
  float v[16], s = 0.0f, ss = 0.0f;
#pragma unroll
  for (int c = 0; c < 16; ++c) { float t = xr[lane + c * 32]; v[c] = t; s += t; ss += t * t; }
  for (int m = 16; m > 0; m >>= 1) { s += __shfl_xor(s, m, 32); ss += __shfl_xor(ss, m, 32); }
  const float mean = s * (1.0f / 512.0f);
  const float var  = ss * (1.0f / 512.0f) - mean * mean;
  const float rstd = rsqrtf(var + 1e-5f);
  float* yr = y + (size_t)wave * D_MODEL;
#pragma unroll
  for (int c = 0; c < 16; ++c) { int d = lane + c * 32; yr[d] = (v[c] - mean) * rstd * g[d] + b[d]; }
}

// ========== Local banded attention (window +-15) with rel-pos bias ==========
// One wave per (b, h, i). q split across lanes: lane holds d=lane and d=lane+32.
__launch_bounds__(256)
__global__ void local_attn_k(const float* __restrict__ Q, const float* __restrict__ Kx,
                             const float* __restrict__ Vx, const float* __restrict__ rel,
                             float* __restrict__ out, int S) {
  const int wave = (blockIdx.x * blockDim.x + threadIdx.x) >> 5;
  const int lane = threadIdx.x & 31;
  const int total = BB * NHEAD * S;
  if (wave >= total) return;
  const int i = wave % S;
  const int h = (wave / S) & (NHEAD - 1);
  const int b = wave / (S * NHEAD);
  const size_t qoff = ((size_t)(b * S + i)) * D_MODEL + h * DHEAD;
  const float q0 = Q[qoff + lane];
  const float q1 = Q[qoff + 32 + lane];

  float myscore = -3.0e38f;
  for (int jj = 0; jj < 2 * WIN + 1; ++jj) {
    const int j = i - WIN + jj;
    if (j < 0 || j >= S) continue;
    const size_t koff = ((size_t)(b * S + j)) * D_MODEL + h * DHEAD;
    float part = (q0 * Kx[koff + lane] + q1 * Kx[koff + 32 + lane]) * 0.125f;
    part += q0 * rel[jj * DHEAD + lane] + q1 * rel[jj * DHEAD + 32 + lane];
    for (int m = 16; m > 0; m >>= 1) part += __shfl_xor(part, m, 32);
    if (lane == jj) myscore = part;
  }
  float mx = myscore;
  for (int m = 16; m > 0; m >>= 1) mx = fmaxf(mx, __shfl_xor(mx, m, 32));
  float w = __expf(myscore - mx);
  if (lane == 31) w = 0.0f;
  float den = w;
  for (int m = 16; m > 0; m >>= 1) den += __shfl_xor(den, m, 32);
  const float inv = 1.0f / den;

  float o0 = 0.0f, o1 = 0.0f;
  for (int jj = 0; jj < 2 * WIN + 1; ++jj) {
    const int j = i - WIN + jj;
    if (j < 0 || j >= S) continue;
    const float wj = __shfl(w, jj, 32) * inv;
    const size_t voff = ((size_t)(b * S + j)) * D_MODEL + h * DHEAD;
    o0 += wj * Vx[voff + lane];
    o1 += wj * Vx[voff + 32 + lane];
  }
  out[qoff + lane] = o0;
  out[qoff + 32 + lane] = o1;
}

// ================= glue kernels =================
__global__ void fill_base_k(float* base) {
  int i = threadIdx.x;
  if (i < 31) base[i] = (float)(i - 15);
}

// X (B,258,512) <- [mu_tok, lv_tok, h(B,256,512)]
__global__ void assemble_enc_k(const float* __restrict__ h, const float* __restrict__ mu_tok,
                               const float* __restrict__ lv_tok, float* __restrict__ X) {
  const int idx = blockIdx.x * 256 + threadIdx.x;
  if (idx >= BB * SENC * D_MODEL) return;
  const int d = idx & (D_MODEL - 1);
  const int row = idx >> 9;
  const int b = row / SENC, t = row % SENC;
  float v;
  if (t == 0) v = mu_tok[d];
  else if (t == 1) v = lv_tok[d];
  else v = h[(((size_t)(b * TT + t - 2)) << 9) + d];
  X[idx] = v;
}

// gather tokens 0 (mu) and 1 (logvar) of each batch -> G (64,512)
__global__ void gather_tokens_k(const float* __restrict__ X, float* __restrict__ G) {
  const int idx = blockIdx.x * 256 + threadIdx.x;
  if (idx >= 64 * D_MODEL) return;
  const int d = idx & (D_MODEL - 1);
  const int r = idx >> 9;
  const int b = r & 31, sel = r >> 5;
  G[idx] = X[(((size_t)(b * SENC + sel)) << 9) + d];
}

// writes mu_r, lv_r into d_out, and masked motion mix for the decoder
__global__ void finalize_latent_k(const float* __restrict__ GO, const float* __restrict__ eps,
                                  const float* __restrict__ motion, float* __restrict__ out,
                                  float* __restrict__ MIX) {
  const int idx = blockIdx.x * 256 + threadIdx.x;
  const int NTOT = BB * TT * D_MOTION;  // 2097152
  if (idx >= NTOT) return;
  const int d = idx & (D_MOTION - 1);
  const int t = (idx >> 8) & (TT - 1);
  const int b = idx >> 16;
  const float mu = GO[(b << 8) + d];
  const float lv = GO[((32 + b) << 8) + d];
  out[NTOT + idx] = mu;          // mu_r
  out[2 * NTOT + idx] = lv;      // lv_r
  const float z = mu + eps[idx] * __expf(0.5f * lv);
  const bool keep = (t < 10) || (t == TT - 1);
  MIX[idx] = keep ? motion[idx] : z;
}

// ================= host orchestration =================
template <int ACT>
static inline void gemm_full_dispatch(hipStream_t st, dim3 grid, const float* A, const float* W,
                                      const float* bias, const float* resid, const float* alpha,
                                      float* C, int M, int N, int K) {
  if (N == 512 && K == 256)
    gemm_wmma_k<ACT, true, 512, 256><<<grid, 256, 0, st>>>(A, W, bias, resid, alpha, C, M, N, K);
  else if (N == 512 && K == 512)
    gemm_wmma_k<ACT, true, 512, 512><<<grid, 256, 0, st>>>(A, W, bias, resid, alpha, C, M, N, K);
  else if (N == 2048 && K == 512)
    gemm_wmma_k<ACT, true, 2048, 512><<<grid, 256, 0, st>>>(A, W, bias, resid, alpha, C, M, N, K);
  else if (N == 512 && K == 2048)
    gemm_wmma_k<ACT, true, 512, 2048><<<grid, 256, 0, st>>>(A, W, bias, resid, alpha, C, M, N, K);
  else if (N == 256 && K == 512)
    gemm_wmma_k<ACT, true, 256, 512><<<grid, 256, 0, st>>>(A, W, bias, resid, alpha, C, M, N, K);
  else
    gemm_wmma_k<ACT, true, 0, 0><<<grid, 256, 0, st>>>(A, W, bias, resid, alpha, C, M, N, K);
}

static inline void gemm(hipStream_t st, const float* A, const float* W, const float* bias,
                        const float* resid, const float* alpha, float* C,
                        int M, int N, int K, int act) {
  dim3 grid((M + BM - 1) / BM, (N + BN - 1) / BN);
  const bool full = (M % BM == 0) && (N % BN == 0) && (K % BK == 0);
  if (full) {
    if (act == ACT_PRELU)      gemm_full_dispatch<ACT_PRELU>(st, grid, A, W, bias, resid, alpha, C, M, N, K);
    else if (act == ACT_RELU)  gemm_full_dispatch<ACT_RELU>(st, grid, A, W, bias, resid, alpha, C, M, N, K);
    else                       gemm_full_dispatch<ACT_NONE>(st, grid, A, W, bias, resid, alpha, C, M, N, K);
  } else {
    if (act == ACT_PRELU)
      gemm_wmma_k<ACT_PRELU, false, 0, 0><<<grid, 256, 0, st>>>(A, W, bias, resid, alpha, C, M, N, K);
    else if (act == ACT_RELU)
      gemm_wmma_k<ACT_RELU, false, 0, 0><<<grid, 256, 0, st>>>(A, W, bias, resid, alpha, C, M, N, K);
    else
      gemm_wmma_k<ACT_NONE, false, 0, 0><<<grid, 256, 0, st>>>(A, W, bias, resid, alpha, C, M, N, K);
  }
}
static inline void lnorm(hipStream_t st, const float* x, const float* g, const float* b,
                         float* y, int rows) {
  layernorm_k<<<(rows + 7) / 8, 256, 0, st>>>(x, g, b, y, rows);
}
static inline void attn(hipStream_t st, const float* Q, const float* K, const float* V,
                        const float* rel, float* O, int S) {
  int waves = BB * NHEAD * S;
  local_attn_k<<<(waves * 32 + 255) / 256, 256, 0, st>>>(Q, K, V, rel, O, S);
}

// d_in flat index map (JAX pytree: dicts flattened in sorted-key order)
#define IDX_EPS 0
#define IDX_MOTION 1
#define DEC_FLN_B 2
#define DEC_FLN_G 3
#define DEC_HEAD_A 4
#define DEC_HEAD_L1W 5
#define DEC_HEAD_L1B 6
#define DEC_HEAD_L2W 7
#define DEC_HEAD_L2B 8
#define DEC_LAYER0 9      /* per layer 26: cross(10) pffn(6) self(10) */
#define DEC_ME 165        /* a1 a2 l1W l1b l2W l2b */
#define DEC_RP 171        /* a l1W l1b l2W l2b */
#define DEC_TE 176
#define ENC_FLN_B 182
#define ENC_FLN_G 183
#define ENC_HEAD_A 184
#define ENC_HEAD_L1W 185
#define ENC_HEAD_L1B 186
#define ENC_HEAD_L2W 187
#define ENC_HEAD_L2B 188
#define ENC_LAYER0 189    /* per layer 16: attn(10) pffn(6) */
#define ENC_LV_TOK 285
#define ENC_ME 286
#define ENC_MU_TOK 292
#define ENC_RP 293
#define IDX_TRAJ 298

extern "C" void kernel_launch(void* const* d_in, const int* in_sizes, int n_in,
                              void* d_out, int out_size, void* d_ws, size_t ws_size,
                              hipStream_t stream) {
  (void)in_sizes; (void)n_in; (void)out_size; (void)ws_size;
  auto F = [&](int i) -> const float* { return (const float*)d_in[i]; };

  const int ME = BB * SENC;   // 8256 encoder tokens
  const int MD = BB * TT;     // 8192 decoder tokens

  // ---- workspace bump allocator ----
  float* ws = (float*)d_ws;
  size_t off = 0;
  auto alloc = [&](size_t n) { float* p = ws + off; off += (n + 63) & ~(size_t)63; return p; };
  float* X    = alloc((size_t)ME * D_MODEL);
  float* LN   = alloc((size_t)ME * D_MODEL);
  float* Qb   = alloc((size_t)ME * D_MODEL);
  float* Kb   = alloc((size_t)ME * D_MODEL);
  float* Vb   = alloc((size_t)ME * D_MODEL);
  float* AT   = alloc((size_t)ME * D_MODEL);
  float* FF   = alloc((size_t)ME * D_FF);
  float* H0   = alloc((size_t)MD * D_MODEL);   // mlp hidden / ctx-LN
  float* CTX  = alloc((size_t)MD * D_MODEL);
  float* MIX  = alloc((size_t)MD * D_MOTION);
  float* RELH = alloc(31 * D_MODEL);
  float* RELT = alloc(31 * DHEAD);
  float* BASE = alloc(64);
  float* GATH = alloc(64 * D_MODEL);
  float* GH   = alloc(64 * D_MODEL);
  float* GO   = alloc(64 * D_MOTION);

  float* out = (float*)d_out;

  fill_base_k<<<1, 32, 0, stream>>>(BASE);

  // ================= ENCODER =================
  gemm(stream, F(IDX_MOTION), F(ENC_ME + 2), F(ENC_ME + 3), nullptr, F(ENC_ME + 0), H0,
       MD, D_MODEL, D_MOTION, ACT_PRELU);
  gemm(stream, H0, F(ENC_ME + 4), F(ENC_ME + 5), nullptr, F(ENC_ME + 1), AT,
       MD, D_MODEL, D_MODEL, ACT_PRELU);
  assemble_enc_k<<<(ME * D_MODEL + 255) / 256, 256, 0, stream>>>(AT, F(ENC_MU_TOK), F(ENC_LV_TOK), X);

  // rel table (central 31 rows only; those are all that the +-15 band touches)
  gemm(stream, BASE, F(ENC_RP + 1), F(ENC_RP + 2), nullptr, F(ENC_RP + 0), RELH, 31, D_MODEL, 1, ACT_PRELU);
  gemm(stream, RELH, F(ENC_RP + 3), F(ENC_RP + 4), nullptr, nullptr, RELT, 31, DHEAD, D_MODEL, ACT_NONE);

  for (int l = 0; l < 6; ++l) {
    const int a = ENC_LAYER0 + 16 * l;       // attn: kW kb lnb lng oW ob qW qb vW vb
    const int p = a + 10;                    // pffn: l1W l1b l2W l2b lnb lng
    lnorm(stream, X, F(a + 3), F(a + 2), LN, ME);
    gemm(stream, LN, F(a + 6), F(a + 7), nullptr, nullptr, Qb, ME, D_MODEL, D_MODEL, ACT_NONE);
    gemm(stream, LN, F(a + 0), F(a + 1), nullptr, nullptr, Kb, ME, D_MODEL, D_MODEL, ACT_NONE);
    gemm(stream, LN, F(a + 8), F(a + 9), nullptr, nullptr, Vb, ME, D_MODEL, D_MODEL, ACT_NONE);
    attn(stream, Qb, Kb, Vb, RELT, AT, SENC);
    gemm(stream, AT, F(a + 4), F(a + 5), X, nullptr, X, ME, D_MODEL, D_MODEL, ACT_NONE);
    lnorm(stream, X, F(p + 5), F(p + 4), LN, ME);
    gemm(stream, LN, F(p + 0), F(p + 1), nullptr, nullptr, FF, ME, D_FF, D_MODEL, ACT_RELU);
    gemm(stream, FF, F(p + 2), F(p + 3), X, nullptr, X, ME, D_MODEL, D_FF, ACT_NONE);
  }
  lnorm(stream, X, F(ENC_FLN_G), F(ENC_FLN_B), LN, ME);
  gather_tokens_k<<<(64 * D_MODEL + 255) / 256, 256, 0, stream>>>(LN, GATH);
  gemm(stream, GATH, F(ENC_HEAD_L1W), F(ENC_HEAD_L1B), nullptr, F(ENC_HEAD_A), GH,
       64, D_MODEL, D_MODEL, ACT_PRELU);
  gemm(stream, GH, F(ENC_HEAD_L2W), F(ENC_HEAD_L2B), nullptr, nullptr, GO,
       64, D_MOTION, D_MODEL, ACT_NONE);
  finalize_latent_k<<<(BB * TT * D_MOTION + 255) / 256, 256, 0, stream>>>(
      GO, F(IDX_EPS), F(IDX_MOTION), out, MIX);

  // ================= DECODER =================
  gemm(stream, MIX, F(DEC_ME + 2), F(DEC_ME + 3), nullptr, F(DEC_ME + 0), H0,
       MD, D_MODEL, D_MOTION, ACT_PRELU);
  gemm(stream, H0, F(DEC_ME + 4), F(DEC_ME + 5), nullptr, F(DEC_ME + 1), X,
       MD, D_MODEL, D_MODEL, ACT_PRELU);
  gemm(stream, F(IDX_TRAJ), F(DEC_TE + 2), F(DEC_TE + 3), nullptr, F(DEC_TE + 0), H0,
       MD, D_MODEL, 3, ACT_PRELU);
  gemm(stream, H0, F(DEC_TE + 4), F(DEC_TE + 5), nullptr, F(DEC_TE + 1), CTX,
       MD, D_MODEL, D_MODEL, ACT_PRELU);
  gemm(stream, BASE, F(DEC_RP + 1), F(DEC_RP + 2), nullptr, F(DEC_RP + 0), RELH, 31, D_MODEL, 1, ACT_PRELU);
  gemm(stream, RELH, F(DEC_RP + 3), F(DEC_RP + 4), nullptr, nullptr, RELT, 31, DHEAD, D_MODEL, ACT_NONE);

  for (int l = 0; l < 6; ++l) {
    const int c = DEC_LAYER0 + 26 * l;  // cross attn params
    const int p = c + 10;               // pffn
    const int s = c + 16;               // self attn
    // self-attention
    lnorm(stream, X, F(s + 3), F(s + 2), LN, MD);
    gemm(stream, LN, F(s + 6), F(s + 7), nullptr, nullptr, Qb, MD, D_MODEL, D_MODEL, ACT_NONE);
    gemm(stream, LN, F(s + 0), F(s + 1), nullptr, nullptr, Kb, MD, D_MODEL, D_MODEL, ACT_NONE);
    gemm(stream, LN, F(s + 8), F(s + 9), nullptr, nullptr, Vb, MD, D_MODEL, D_MODEL, ACT_NONE);
    attn(stream, Qb, Kb, Vb, RELT, AT, TT);
    gemm(stream, AT, F(s + 4), F(s + 5), X, nullptr, X, MD, D_MODEL, D_MODEL, ACT_NONE);
    // cross-attention (LN applied to both x and ctx with this block's params)
    lnorm(stream, X, F(c + 3), F(c + 2), LN, MD);
    lnorm(stream, CTX, F(c + 3), F(c + 2), H0, MD);
    gemm(stream, LN, F(c + 6), F(c + 7), nullptr, nullptr, Qb, MD, D_MODEL, D_MODEL, ACT_NONE);
    gemm(stream, H0, F(c + 0), F(c + 1), nullptr, nullptr, Kb, MD, D_MODEL, D_MODEL, ACT_NONE);
    gemm(stream, H0, F(c + 8), F(c + 9), nullptr, nullptr, Vb, MD, D_MODEL, D_MODEL, ACT_NONE);
    attn(stream, Qb, Kb, Vb, RELT, AT, TT);
    gemm(stream, AT, F(c + 4), F(c + 5), X, nullptr, X, MD, D_MODEL, D_MODEL, ACT_NONE);
    // pffn
    lnorm(stream, X, F(p + 5), F(p + 4), LN, MD);
    gemm(stream, LN, F(p + 0), F(p + 1), nullptr, nullptr, FF, MD, D_FF, D_MODEL, ACT_RELU);
    gemm(stream, FF, F(p + 2), F(p + 3), X, nullptr, X, MD, D_MODEL, D_FF, ACT_NONE);
  }
  lnorm(stream, X, F(DEC_FLN_G), F(DEC_FLN_B), LN, MD);
  gemm(stream, LN, F(DEC_HEAD_L1W), F(DEC_HEAD_L1B), nullptr, F(DEC_HEAD_A), AT,
       MD, D_MODEL, D_MODEL, ACT_PRELU);
  gemm(stream, AT, F(DEC_HEAD_L2W), F(DEC_HEAD_L2B), nullptr, nullptr, out,
       MD, D_MOTION, D_MODEL, ACT_NONE);
}
